// BasicTransformerBlock_57586921504867
// MI455X (gfx1250) — matrix operand order repaired and checked by reading, not compiled, not run on hardware
//
#include <hip/hip_runtime.h>
#include <cstddef>
#include <cstdint>

#define DIM        512
#define HEADS      8
#define DHEAD      64
#define BATCH      8
#define SEQ        1024
#define NTOK       8192          /* 8*1024 */
#define XDIM       768           /* cross dim */
#define XSEQ       77
#define FFI        2048          /* FF inner */
#define SCALE_ATTN 0.125f        /* 64^-0.5 */

typedef __bf16 bf16;
typedef bf16  v16bf __attribute__((ext_vector_type(16)));
typedef float v8f   __attribute__((ext_vector_type(8)));
typedef unsigned int u32x4 __attribute__((ext_vector_type(4)));

union BF16x16 { v16bf v; unsigned short s[16]; u32x4 q[2]; };

__device__ __forceinline__ unsigned short f2bfu(float f) {
  unsigned u = __float_as_uint(f);
  unsigned r = u + 0x7FFFu + ((u >> 16) & 1u);  // round-to-nearest-even
  return (unsigned short)(r >> 16);
}
__device__ __forceinline__ v8f wmma_bf16(v16bf a, v16bf b, v8f c) {
  // 8 args: (neg_a, A, neg_b, B, c_mod, C, reuse_a, reuse_b)
  return __builtin_amdgcn_wmma_f32_16x16x32_bf16(false, a, false, b, (short)0, c,
                                                 false, false);
}
__device__ __forceinline__ float gelu_exact(float x) {
  return 0.5f * x * (1.0f + erff(x * 0.70710678118654752f));
}

// CDNA5 async memory->LDS copy (VGLOBAL encoding, GV mode, ASYNCcnt-tracked).
// Low 32 bits of a generic LDS pointer are the in-wave LDS byte offset
__device__ __forceinline__ void async_ld_b128(void* lds, const void* g) {
  asm volatile("global_load_async_to_lds_b128 %0, %1, off"
               :: "v"((unsigned)(uintptr_t)lds),
                  "v"((unsigned long long)(uintptr_t)g)
               : "memory");
}
#define WAIT_ASYNCCNT(N) asm volatile("s_wait_asynccnt " #N ::: "memory")

// ---------------------------------------------------------------- cast f32->bf16
__global__ __launch_bounds__(256)
void cast_bf_kernel(const float* __restrict__ src, unsigned short* __restrict__ dst,
                    int n) {
  int i = blockIdx.x * 256 + threadIdx.x;
  if (i < n) dst[i] = f2bfu(src[i]);
}

// ------------------------------------------------- weight transpose + cast
// src: [K][N] f32 row-major -> dst: [N][K] bf16 row-major. K,N multiples of 32.
__global__ __launch_bounds__(256)
void cast_transpose_kernel(const float* __restrict__ src,
                           unsigned short* __restrict__ dst, int K, int N) {
  __shared__ float tile[32][33];
  int kb = blockIdx.x * 32, nb = blockIdx.y * 32;
  int tx = threadIdx.x & 31, ty = threadIdx.x >> 5;   // 32 x 8
#pragma unroll
  for (int r = 0; r < 32; r += 8)
    tile[ty + r][tx] = src[(size_t)(kb + ty + r) * N + nb + tx];
  __syncthreads();
#pragma unroll
  for (int r = 0; r < 32; r += 8)
    dst[(size_t)(nb + ty + r) * K + kb + tx] = f2bfu(tile[tx][ty + r]);
}

// ---------------------------------------------------------------- layernorm + cast
__global__ __launch_bounds__(256)
void ln_cast_kernel(const float* __restrict__ x, const float* __restrict__ g,
                    const float* __restrict__ b, unsigned short* __restrict__ out) {
  int row = blockIdx.x;
  int t = threadIdx.x;
  const float* xr = x + (size_t)row * DIM;
  float v0 = xr[t], v1 = xr[t + 256];
  __shared__ float red[256];
  red[t] = v0 + v1;
  __syncthreads();
  for (int o = 128; o > 0; o >>= 1) { if (t < o) red[t] += red[t + o]; __syncthreads(); }
  float mu = red[0] * (1.0f / DIM);
  __syncthreads();
  float d0 = v0 - mu, d1 = v1 - mu;
  red[t] = d0 * d0 + d1 * d1;
  __syncthreads();
  for (int o = 128; o > 0; o >>= 1) { if (t < o) red[t] += red[t + o]; __syncthreads(); }
  float rstd = rsqrtf(red[0] * (1.0f / DIM) + 1e-5f);
  unsigned short* orow = out + (size_t)row * DIM;
  orow[t]       = f2bfu(d0 * rstd * g[t]       + b[t]);
  orow[t + 256] = f2bfu(d1 * rstd * g[t + 256] + b[t + 256]);
}

// ---------------------------------------------------------------- generic bf16 GEMM
// C[M,N] = A[M,K] * WT[N,K]^T (+bias). mode 0: store bf16. mode 1: outF += result.
// BM=BN=64, BK=32, 128 threads (4 waves). Async double-buffered LDS staging.
__global__ __launch_bounds__(128)
void gemm_bf16_kernel(const unsigned short* __restrict__ A, int lda,
                      const unsigned short* __restrict__ WT, int ldwt,
                      const float* __restrict__ bias,
                      int M, int K, int mode,
                      unsigned short* __restrict__ outBf,
                      float* __restrict__ outF, int ldo) {
  __shared__ unsigned short As[2][64 * 32];   // [m][k]
  __shared__ unsigned short Bs[2][64 * 32];   // [n][k]
  int tid = threadIdx.x;
  int lane = tid & 31, w = tid >> 5;
  int n0 = blockIdx.x * 64;
  int m0 = blockIdx.y * 64;
  int lr = lane & 15;
  int hi = (lane >> 4) & 1;
  int kbA = hi ? 8 : 0;    // unused: A frag is loaded like B, elems 0..15 -> k=kbB..+15
  int kbB = hi ? 16 : 0;   // B frag: elems 0..15 -> k=kbB..+15

  v8f acc[4];
  for (int i = 0; i < 4; ++i)
    for (int j = 0; j < 8; ++j) acc[i][j] = 0.f;

  // each thread owns 2 b128 slots per tile: idx = tid, tid+128
  auto issue = [&](int buf, int kt) {
    int k0 = kt << 5;
#pragma unroll
    for (int r = 0; r < 2; ++r) {
      int idx = tid + r * 128;
      int row = idx >> 2, q = idx & 3;          // row 0..63, q*8 within 32 k's
      int gm = m0 + row; gm = gm < M ? gm : M - 1;   // clamp: junk rows discarded
      async_ld_b128(&As[buf][row * 32 + q * 8],
                    A + (size_t)gm * lda + k0 + q * 8);
      async_ld_b128(&Bs[buf][row * 32 + q * 8],
                    WT + (size_t)(n0 + row) * ldwt + k0 + q * 8);
    }
  };

  int nk = K >> 5;
  issue(0, 0);
  for (int kt = 0; kt < nk; ++kt) {
    if (kt + 1 < nk) { issue((kt + 1) & 1, kt + 1); WAIT_ASYNCCNT(4); }
    else             { WAIT_ASYNCCNT(0); }
    __syncthreads();
    int buf = kt & 1;
    BF16x16 fa;
    const unsigned short* arow = &As[buf][(w * 16 + lr) * 32];
    fa.q[0] = *(const u32x4*)(arow + kbB);
    fa.q[1] = *(const u32x4*)(arow + kbB + 8);
#pragma unroll
    for (int nf = 0; nf < 4; ++nf) {
      BF16x16 fb;
      const unsigned short* brow = &Bs[buf][(nf * 16 + lr) * 32];
      fb.q[0] = *(const u32x4*)(brow + kbB);
      fb.q[1] = *(const u32x4*)(brow + kbB + 8);
      acc[nf] = wmma_bf16(fa.v, fb.v, acc[nf]);
    }
    __syncthreads();
  }

  int rbase = m0 + w * 16 + (hi ? 8 : 0);   // C frag: vgpr j -> row j (+8 hi lanes)
#pragma unroll
  for (int nf = 0; nf < 4; ++nf) {
    int gn = n0 + nf * 16 + lr;
    float bv = bias ? bias[gn] : 0.f;
#pragma unroll
    for (int j = 0; j < 8; ++j) {
      int gm = rbase + j;
      if (gm < M) {
        float r = acc[nf][j] + bv;
        if (mode == 0) outBf[(size_t)gm * ldo + gn] = f2bfu(r);
        else           outF[(size_t)gm * ldo + gn] += r;
      }
    }
  }
}

// ---------------------------------------------------------------- GEGLU fused GEMM
// out[m][n] = (A.WT[n] + bg[n]) * gelu(A.WT[n+Ninner] + bg[n+Ninner]); WT is [2*Ninner][K]
__global__ __launch_bounds__(128)
void gemm_geglu_kernel(const unsigned short* __restrict__ A, int lda,
                       const unsigned short* __restrict__ WT, int ldwt,
                       const float* __restrict__ bg,
                       int M, int K, int Ninner,
                       unsigned short* __restrict__ out, int ldo) {
  __shared__ unsigned short As[2][64 * 32];
  __shared__ unsigned short Bu[2][64 * 32];
  __shared__ unsigned short Bg[2][64 * 32];
  int tid = threadIdx.x;
  int lane = tid & 31, w = tid >> 5;
  int n0 = blockIdx.x * 64;
  int m0 = blockIdx.y * 64;
  int lr = lane & 15;
  int hi = (lane >> 4) & 1;
  int kbA = hi ? 8 : 0;
  int kbB = hi ? 16 : 0;

  v8f accU[4], accG[4];
  for (int i = 0; i < 4; ++i)
    for (int j = 0; j < 8; ++j) { accU[i][j] = 0.f; accG[i][j] = 0.f; }

  auto issue = [&](int buf, int kt) {
    int k0 = kt << 5;
#pragma unroll
    for (int r = 0; r < 2; ++r) {
      int idx = tid + r * 128;
      int row = idx >> 2, q = idx & 3;
      int gm = m0 + row; gm = gm < M ? gm : M - 1;
      async_ld_b128(&As[buf][row * 32 + q * 8],
                    A + (size_t)gm * lda + k0 + q * 8);
      async_ld_b128(&Bu[buf][row * 32 + q * 8],
                    WT + (size_t)(n0 + row) * ldwt + k0 + q * 8);
      async_ld_b128(&Bg[buf][row * 32 + q * 8],
                    WT + (size_t)(n0 + Ninner + row) * ldwt + k0 + q * 8);
    }
  };

  int nk = K >> 5;
  issue(0, 0);
  for (int kt = 0; kt < nk; ++kt) {
    if (kt + 1 < nk) { issue((kt + 1) & 1, kt + 1); WAIT_ASYNCCNT(6); }
    else             { WAIT_ASYNCCNT(0); }
    __syncthreads();
    int buf = kt & 1;
    BF16x16 fa;
    const unsigned short* arow = &As[buf][(w * 16 + lr) * 32];
    fa.q[0] = *(const u32x4*)(arow + kbB);
    fa.q[1] = *(const u32x4*)(arow + kbB + 8);
#pragma unroll
    for (int nf = 0; nf < 4; ++nf) {
      BF16x16 fu, fg;
      const unsigned short* burow = &Bu[buf][(nf * 16 + lr) * 32];
      const unsigned short* bgrow = &Bg[buf][(nf * 16 + lr) * 32];
      fu.q[0] = *(const u32x4*)(burow + kbB);
      fu.q[1] = *(const u32x4*)(burow + kbB + 8);
      fg.q[0] = *(const u32x4*)(bgrow + kbB);
      fg.q[1] = *(const u32x4*)(bgrow + kbB + 8);
      accU[nf] = wmma_bf16(fa.v, fu.v, accU[nf]);
      accG[nf] = wmma_bf16(fa.v, fg.v, accG[nf]);
    }
    __syncthreads();
  }

  int rbase = m0 + w * 16 + (hi ? 8 : 0);
#pragma unroll
  for (int nf = 0; nf < 4; ++nf) {
    int gn = n0 + nf * 16 + lr;
    float bu = bg[gn];
    float bgv = bg[gn + Ninner];
#pragma unroll
    for (int j = 0; j < 8; ++j) {
      int gm = rbase + j;
      if (gm < M) {
        float u = accU[nf][j] + bu;
        float g = accG[nf][j] + bgv;
        out[(size_t)gm * ldo + gn] = f2bfu(u * gelu_exact(g));
      }
    }
  }
}

// ---------------------------------------------------------------- fused attention
// grid: (Sq/64, BATCH*HEADS), 128 threads (4 waves), wave owns 16 q rows.
__global__ __launch_bounds__(128)
void attn_kernel(const unsigned short* __restrict__ Q,
                 const unsigned short* __restrict__ Kb,
                 const unsigned short* __restrict__ Vb,
                 unsigned short* __restrict__ O,
                 int Sq, int Sk) {
  __shared__ unsigned short Kls[2][32 * 64];   // [key][d]
  __shared__ unsigned short Vls[2][32 * 64];   // [key][d]
  __shared__ unsigned short Pls[4][16 * 32];   // per-wave P tile [m][key]
  int tid = threadIdx.x;
  int lane = tid & 31, w = tid >> 5;
  int bh = blockIdx.y;
  int b = bh >> 3, h = bh & 7;
  int q0 = blockIdx.x * 64;
  int lr = lane & 15;
  int hi = (lane >> 4) & 1;
  int hl = hi ? 8 : 0;
  int kbA = hi ? 8 : 0;
  int kbB = hi ? 16 : 0;

  const unsigned short* Qbase = Q  + ((size_t)(b * Sq + q0)) * DIM + h * DHEAD;
  const unsigned short* Kbase = Kb + ((size_t)(b * Sk)) * DIM + h * DHEAD;
  const unsigned short* Vbase = Vb + ((size_t)(b * Sk)) * DIM + h * DHEAD;

  auto issue = [&](int buf, int kt) {
    int kbase = kt << 5;
#pragma unroll
    for (int r = 0; r < 2; ++r) {
      int idx = tid + r * 128;
      int row = idx >> 3, q = idx & 7;           // row 0..31, q*8 within 64 d's
      int gk = kbase + row; gk = gk < Sk ? gk : Sk - 1;  // clamp: masked keys -> p=0
      async_ld_b128(&Kls[buf][row * 64 + q * 8],
                    Kbase + (size_t)gk * DIM + q * 8);
      async_ld_b128(&Vls[buf][row * 64 + q * 8],
                    Vbase + (size_t)gk * DIM + q * 8);
    }
  };

  // Q fragments for this wave's 16 rows (d = 64 -> 2 frags)
  BF16x16 qf[2];
  const unsigned short* qrow = Qbase + (size_t)(w * 16 + lr) * DIM;
#pragma unroll
  for (int f = 0; f < 2; ++f) {
    qf[f].q[0] = *(const u32x4*)(qrow + f * 32 + kbB);
    qf[f].q[1] = *(const u32x4*)(qrow + f * 32 + kbB + 8);
  }

  float mrow[8], lrow[8];
#pragma unroll
  for (int j = 0; j < 8; ++j) { mrow[j] = -3.0e38f; lrow[j] = 0.f; }
  v8f oacc[4];
  for (int i = 0; i < 4; ++i)
    for (int j = 0; j < 8; ++j) oacc[i][j] = 0.f;

  int nkt = (Sk + 31) >> 5;
  issue(0, 0);
  for (int kt = 0; kt < nkt; ++kt) {
    if (kt + 1 < nkt) { issue((kt + 1) & 1, kt + 1); WAIT_ASYNCCNT(4); }
    else              { WAIT_ASYNCCNT(0); }
    __syncthreads();
    int buf = kt & 1;
    int kbase = kt << 5;

    // scores: two 16x16 frags over 32 keys, chained over d = 0..63
    v8f sfr[2];
#pragma unroll
    for (int nt = 0; nt < 2; ++nt) {
      v8f s;
      for (int j = 0; j < 8; ++j) s[j] = 0.f;
      int n = nt * 16 + lr;   // key within tile
#pragma unroll
      for (int f = 0; f < 2; ++f) {
        BF16x16 fb;
        const unsigned short* krow = &Kls[buf][n * 64 + f * 32];
        fb.q[0] = *(const u32x4*)(krow + kbB);
        fb.q[1] = *(const u32x4*)(krow + kbB + 8);
        s = wmma_bf16(qf[f].v, fb.v, s);
      }
      sfr[nt] = s;
    }

    // online softmax (rows j+hl; reductions within 16-lane half-groups)
    int key0 = kbase + lr, key1 = kbase + 16 + lr;
#pragma unroll
    for (int j = 0; j < 8; ++j) {
      float s0 = sfr[0][j] * SCALE_ATTN;
      float s1 = sfr[1][j] * SCALE_ATTN;
      if (key0 >= Sk) s0 = -3.0e38f;
      if (key1 >= Sk) s1 = -3.0e38f;
      float m = fmaxf(s0, s1);
      for (int o = 1; o < 16; o <<= 1) m = fmaxf(m, __shfl_xor(m, o, 16));
      float mn = fmaxf(mrow[j], m);
      float alpha = __expf(mrow[j] - mn);
      mrow[j] = mn;
      float p0 = __expf(s0 - mn);
      float p1 = __expf(s1 - mn);
      float sum = p0 + p1;
      for (int o = 1; o < 16; o <<= 1) sum += __shfl_xor(sum, o, 16);
      lrow[j] = lrow[j] * alpha + sum;
#pragma unroll
      for (int dt = 0; dt < 4; ++dt) oacc[dt][j] *= alpha;
      Pls[w][(j + hl) * 32 + lr]      = f2bfu(p0);
      Pls[w][(j + hl) * 32 + 16 + lr] = f2bfu(p1);
    }

    // P is a 16x32 A-fragment over the 32 keys
    BF16x16 pf;
    const unsigned short* prow = &Pls[w][lr * 32];
    pf.q[0] = *(const u32x4*)(prow + kbB);
    pf.q[1] = *(const u32x4*)(prow + kbB + 8);

    // O += P * V   (4 x 16-col head-dim tiles)
#pragma unroll
    for (int dt = 0; dt < 4; ++dt) {
      BF16x16 vf;
      int n = dt * 16 + lr;   // head-dim column
#pragma unroll
      for (int e = 0; e < 16; ++e) vf.s[e] = Vls[buf][(kbB + e) * 64 + n];
      oacc[dt] = wmma_bf16(pf.v, vf.v, oacc[dt]);
    }
    __syncthreads();
  }

  unsigned short* Obase = O + ((size_t)(b * Sq + q0 + w * 16)) * DIM + h * DHEAD;
#pragma unroll
  for (int dt = 0; dt < 4; ++dt) {
#pragma unroll
    for (int j = 0; j < 8; ++j) {
      float val = oacc[dt][j] / lrow[j];
      Obase[(size_t)(j + hl) * DIM + dt * 16 + lr] = f2bfu(val);
    }
  }
}

// ================================================================ host side
extern "C" void kernel_launch(void* const* d_in, const int* in_sizes, int n_in,
                              void* d_out, int out_size, void* d_ws, size_t ws_size,
                              hipStream_t stream) {
  (void)in_sizes; (void)n_in; (void)out_size; (void)ws_size;
  const float* x    = (const float*)d_in[0];
  const float* enc  = (const float*)d_in[1];
  const float* ln1g = (const float*)d_in[2];
  const float* ln1b = (const float*)d_in[3];
  const float* wq1  = (const float*)d_in[4];
  const float* wk1  = (const float*)d_in[5];
  const float* wv1  = (const float*)d_in[6];
  const float* wo1  = (const float*)d_in[7];
  const float* bo1  = (const float*)d_in[8];
  const float* ln2g = (const float*)d_in[9];
  const float* ln2b = (const float*)d_in[10];
  const float* wq2  = (const float*)d_in[11];
  const float* wk2  = (const float*)d_in[12];
  const float* wv2  = (const float*)d_in[13];
  const float* wo2  = (const float*)d_in[14];
  const float* bo2  = (const float*)d_in[15];
  const float* ln3g = (const float*)d_in[16];
  const float* ln3b = (const float*)d_in[17];
  const float* wg   = (const float*)d_in[18];
  const float* bg   = (const float*)d_in[19];
  const float* wf   = (const float*)d_in[20];
  const float* bfb  = (const float*)d_in[21];
  float* out = (float*)d_out;

  char* ws = (char*)d_ws;
  size_t off = 0;
  auto alloc = [&](size_t bytes) -> void* {
    size_t o = off;
    off += (bytes + 255) & ~(size_t)255;
    return (void*)(ws + o);
  };
  unsigned short* h_bf    = (unsigned short*)alloc((size_t)NTOK * DIM * 2);
  unsigned short* q_bf    = (unsigned short*)alloc((size_t)NTOK * DIM * 2);
  unsigned short* k_bf    = (unsigned short*)alloc((size_t)NTOK * DIM * 2);
  unsigned short* v_bf    = (unsigned short*)alloc((size_t)NTOK * DIM * 2);
  unsigned short* attn_bf = (unsigned short*)alloc((size_t)NTOK * DIM * 2);
  unsigned short* ffi_bf  = (unsigned short*)alloc((size_t)NTOK * FFI * 2);
  // transposed bf16 weights: [N][K]
  unsigned short* wq1_t   = (unsigned short*)alloc((size_t)DIM * DIM * 2);
  unsigned short* wk1_t   = (unsigned short*)alloc((size_t)DIM * DIM * 2);
  unsigned short* wv1_t   = (unsigned short*)alloc((size_t)DIM * DIM * 2);
  unsigned short* wo1_t   = (unsigned short*)alloc((size_t)DIM * DIM * 2);
  unsigned short* wq2_t   = (unsigned short*)alloc((size_t)DIM * DIM * 2);
  unsigned short* wk2_t   = (unsigned short*)alloc((size_t)XDIM * DIM * 2);
  unsigned short* wv2_t   = (unsigned short*)alloc((size_t)XDIM * DIM * 2);
  unsigned short* wo2_t   = (unsigned short*)alloc((size_t)DIM * DIM * 2);
  unsigned short* wg_t    = (unsigned short*)alloc((size_t)DIM * 2 * FFI * 2);
  unsigned short* wf_t    = (unsigned short*)alloc((size_t)FFI * DIM * 2);
  unsigned short* enc_bf  = (unsigned short*)alloc((size_t)BATCH * XSEQ * XDIM * 2);

  auto castT = [&](const float* src, unsigned short* dst, int K_, int N_) {
    cast_transpose_kernel<<<dim3(K_ / 32, N_ / 32), dim3(256), 0, stream>>>(
        src, dst, K_, N_);
  };
  auto gemm = [&](const unsigned short* A, int lda, const unsigned short* WT,
                  int ldwt, const float* bias, int M, int N, int K, int mode,
                  unsigned short* ob, float* of, int ldo) {
    gemm_bf16_kernel<<<dim3(N / 64, (M + 63) / 64), dim3(128), 0, stream>>>(
        A, lda, WT, ldwt, bias, M, K, mode, ob, of, ldo);
  };

  // residual accumulator = x
  hipMemcpyAsync(out, x, (size_t)NTOK * DIM * sizeof(float),
                 hipMemcpyDeviceToDevice, stream);

  // weight transposes + enc cast
  castT(wq1, wq1_t, DIM, DIM);
  castT(wk1, wk1_t, DIM, DIM);
  castT(wv1, wv1_t, DIM, DIM);
  castT(wo1, wo1_t, DIM, DIM);
  castT(wq2, wq2_t, DIM, DIM);
  castT(wk2, wk2_t, XDIM, DIM);
  castT(wv2, wv2_t, XDIM, DIM);
  castT(wo2, wo2_t, DIM, DIM);
  castT(wg,  wg_t,  DIM, 2 * FFI);
  castT(wf,  wf_t,  FFI, DIM);
  cast_bf_kernel<<<dim3((BATCH * XSEQ * XDIM + 255) / 256), dim3(256), 0, stream>>>(
      enc, enc_bf, BATCH * XSEQ * XDIM);

  // ---- self-attention block
  ln_cast_kernel<<<dim3(NTOK), dim3(256), 0, stream>>>(out, ln1g, ln1b, h_bf);
  gemm(h_bf, DIM, wq1_t, DIM, nullptr, NTOK, DIM, DIM, 0, q_bf, nullptr, DIM);
  gemm(h_bf, DIM, wk1_t, DIM, nullptr, NTOK, DIM, DIM, 0, k_bf, nullptr, DIM);
  gemm(h_bf, DIM, wv1_t, DIM, nullptr, NTOK, DIM, DIM, 0, v_bf, nullptr, DIM);
  attn_kernel<<<dim3(SEQ / 64, BATCH * HEADS), dim3(128), 0, stream>>>(
      q_bf, k_bf, v_bf, attn_bf, SEQ, SEQ);
  gemm(attn_bf, DIM, wo1_t, DIM, bo1, NTOK, DIM, DIM, 1, nullptr, out, DIM);

  // ---- cross-attention block
  ln_cast_kernel<<<dim3(NTOK), dim3(256), 0, stream>>>(out, ln2g, ln2b, h_bf);
  gemm(h_bf, DIM, wq2_t, DIM, nullptr, NTOK, DIM, DIM, 0, q_bf, nullptr, DIM);
  gemm(enc_bf, XDIM, wk2_t, XDIM, nullptr, BATCH * XSEQ, DIM, XDIM, 0, k_bf, nullptr, DIM);
  gemm(enc_bf, XDIM, wv2_t, XDIM, nullptr, BATCH * XSEQ, DIM, XDIM, 0, v_bf, nullptr, DIM);
  attn_kernel<<<dim3(SEQ / 64, BATCH * HEADS), dim3(128), 0, stream>>>(
      q_bf, k_bf, v_bf, attn_bf, SEQ, XSEQ);
  gemm(attn_bf, DIM, wo2_t, DIM, bo2, NTOK, DIM, DIM, 1, nullptr, out, DIM);

  // ---- GEGLU feed-forward block
  ln_cast_kernel<<<dim3(NTOK), dim3(256), 0, stream>>>(out, ln3g, ln3b, h_bf);
  gemm_geglu_kernel<<<dim3(FFI / 64, NTOK / 64), dim3(128), 0, stream>>>(
      h_bf, DIM, wg_t, DIM, bg, NTOK, DIM, FFI, ffi_bf, FFI);
  gemm(ffi_bf, FFI, wf_t, FFI, bfb, NTOK, DIM, FFI, 1, nullptr, out, DIM);
}
